// Quantizer_impl_19731079757831
// MI455X (gfx1250) — matrix operand order repaired and checked by reading, not compiled, not run on hardware
//
#include <hip/hip_runtime.h>
#include <hip/hip_bf16.h>
#include <stdint.h>

typedef __attribute__((ext_vector_type(16))) __bf16 v16bf;
typedef __attribute__((ext_vector_type(8)))  __bf16 v8bf;
typedef __attribute__((ext_vector_type(8)))  float  v8f;

#define NB    32
#define CD    256
#define HWSZ  1024
#define NPIX  32768      // 32 * 32 * 32
#define NE    2048
#define KAUG  288        // 256 channels + 2 norm channels + pad to 9*32
#define KCH   9          // KAUG / 32
#define TILEW (16 * KAUG)        // ushorts per 16-row bf16 tile
#define TILEB (16 * KAUG * 2)    // bytes per tile = 9216

__device__ __forceinline__ unsigned short f2bf(float f) {
    union { float f; unsigned u; } v; v.f = f;
    unsigned u = v.u;
    u += 0x7FFFu + ((u >> 16) & 1u);   // round-to-nearest-even
    return (unsigned short)(u >> 16);
}
__device__ __forceinline__ float bf2f(unsigned short h) {
    union { unsigned u; float f; } v; v.u = ((unsigned)h) << 16;
    return v.f;
}

// ---------------------------------------------------------------------------
// Kernel 1: build augmented bf16 codebook  Waug[code][0..255]=bf16(w),
// [256..257] = two-term bf16 split of -0.5*||w||^2, [258..287]=0.
// Also zeroes the loss accumulator.
// ---------------------------------------------------------------------------
__global__ __launch_bounds__(256)
void vq_prep(const float* __restrict__ w, unsigned short* __restrict__ waug,
             float* __restrict__ loss_acc) {
    __shared__ float red[256];
    const int code = blockIdx.x;
    const int t = threadIdx.x;
    float v = w[code * CD + t];
    waug[code * KAUG + t] = f2bf(v);
    red[t] = v * v;
    __syncthreads();
    for (int s = 128; s > 0; s >>= 1) {
        if (t < s) red[t] += red[t + s];
        __syncthreads();
    }
    if (t == 0) {
        float nh = -0.5f * red[0];
        unsigned short hi = f2bf(nh);
        unsigned short lo = f2bf(nh - bf2f(hi));
        waug[code * KAUG + 256] = hi;
        waug[code * KAUG + 257] = lo;
        if (code == 0) *loss_acc = 0.0f;
    }
    if (t >= 2 && t < 32) waug[code * KAUG + 256 + t] = 0;  // pad channels
}

// ---------------------------------------------------------------------------
// Kernel 2: fused distance-GEMM + argmax via WMMA bf16.
// Block = 128 threads = 4 waves. Each wave owns TWO 16-pixel tiles as
// persistent B fragments (2 WMMAs per A-fragment load -> halves both the
// L2 codebook traffic and the per-WMMA ds_load stall). Codebook tiles are
// double-buffer staged with GLOBAL_LOAD_ASYNC_TO_LDS (ASYNCcnt DMA).
// The single shared buffer is reused: x-staging (2 rounds) then code tiles.
// argmin(dist) == argmax(x.w - 0.5||w||^2).
// ---------------------------------------------------------------------------
__global__ __launch_bounds__(128)
void vq_argmin(const float* __restrict__ x,
               const unsigned short* __restrict__ waug,
               int* __restrict__ idx_ws, int* __restrict__ idx_out) {
    // 36,864 B: holds 4 pixel tiles during staging, then 2 async code tiles.
    __shared__ alignas(32) unsigned short sbuf[4 * TILEW];

    const int tid  = threadIdx.x;
    const int lane = tid & 31;
    const int wv   = tid >> 5;

    const int blockBase = blockIdx.x * 128;   // 128 pixels, single image b
    const int b = blockBase >> 10;

    const int ncol = lane & 15;
    const int kb   = (lane >> 4) * 16;        // B frag: lanes 16-31 = K+16
    const int m    = lane & 15;               // A row (code within tile)
    const int koff = (lane >> 4) * 16;        // A layout half-wave K offset

    v16bf Breg0[KCH], Breg1[KCH];

    // ---- two staging rounds: 64 pixels x 288 aug channels each ----
    #pragma unroll
    for (int rnd = 0; rnd < 2; ++rnd) {
        const int hw0 = (blockBase & 1023) + rnd * 64;
        {
            const int p  = tid & 63;
            const int c0 = tid >> 6;          // 0 or 1
            const float* xb = x + ((size_t)b << 18) + (size_t)(hw0 + p);
            unsigned short* dst = sbuf + (size_t)p * KAUG;
            #pragma unroll 4
            for (int j = 0; j < 144; ++j) {
                int c = c0 + 2 * j;
                unsigned short hv;
                if (c < 256)      hv = f2bf(xb[(size_t)c << 10]);
                else if (c < 258) hv = 0x3F80;  // bf16(1.0): norm term pickup
                else              hv = 0;
                dst[c] = hv;
            }
        }
        __syncthreads();
        const unsigned short* myTile = sbuf + wv * TILEW;
        #pragma unroll
        for (int ch = 0; ch < KCH; ++ch) {
            v16bf f = *(const v16bf*)(myTile + ncol * KAUG + ch * 32 + kb);
            if (rnd == 0) Breg0[ch] = f; else Breg1[ch] = f;
        }
        __syncthreads();   // done reading before buffer reuse
    }

    // ---- async-DMA prologue: stage code tile 0 into buffer 0 ----
    // Low 32 bits of a generic LDS pointer are the LDS byte address.
    const unsigned ldsA = (unsigned)(size_t)(const void*)sbuf;
    if (wv == 0) {
        unsigned dst = ldsA + lane * 16;       // 16B chunk per lane
        int      off = lane * 16;              // byte offset into waug
        #pragma unroll
        for (int j = 0; j < 18; ++j) {         // 18 * 32 lanes * 16B = 9216B
            asm volatile("global_load_async_to_lds_b128 %0, %1, %2"
                         :: "v"(dst + j * 512), "v"(off + j * 512), "s"(waug)
                         : "memory");
        }
        asm volatile("s_wait_asynccnt 0x0" ::: "memory");
    }
    __syncthreads();

    float bestV0 = -3.0e38f, bestV1 = -3.0e38f;
    int   bestI0 = 0,        bestI1 = 0;

    for (int ct = 0; ct < NE / 16; ++ct) {
        const int buf = ct & 1;

        // issue async stage of tile ct+1 into the other buffer (overlapped)
        if (wv == 0 && ct + 1 < NE / 16) {
            unsigned dst = ldsA + (buf ^ 1) * TILEB + lane * 16;
            int      off = (ct + 1) * TILEB + lane * 16;
            #pragma unroll
            for (int j = 0; j < 18; ++j) {
                asm volatile("global_load_async_to_lds_b128 %0, %1, %2"
                             :: "v"(dst + j * 512), "v"(off + j * 512), "s"(waug)
                             : "memory");
            }
        }

        // each A fragment feeds two WMMAs (both pixel tiles)
        const unsigned short* ar0 = sbuf + buf * TILEW + m * KAUG + koff;
        v8f acc0 = {}, acc1 = {};
        #pragma unroll
        for (int ch = 0; ch < KCH; ++ch) {
            union { v16bf v; v8bf h[2]; } A;
            A.h[0] = *(const v8bf*)(ar0 + ch * 32);        // K {0..7}
            A.h[1] = *(const v8bf*)(ar0 + ch * 32 + 8);    // K {8..15}
            acc0 = __builtin_amdgcn_wmma_f32_16x16x32_bf16(
                false, A.v, false, Breg0[ch], (short)0, acc0, false, false);
            acc1 = __builtin_amdgcn_wmma_f32_16x16x32_bf16(
                false, A.v, false, Breg1[ch], (short)0, acc1, false, false);
        }

        const int cbase = ct * 16 + (lane >> 4) * 8;
        #pragma unroll
        for (int r = 0; r < 8; ++r) {
            float s0 = acc0[r];
            if (s0 > bestV0) { bestV0 = s0; bestI0 = cbase + r; }
            float s1 = acc1[r];
            if (s1 > bestV1) { bestV1 = s1; bestI1 = cbase + r; }
        }

        if (wv == 0)
            asm volatile("s_wait_asynccnt 0x0" ::: "memory");
        __syncthreads();   // next tile's DMA complete + this buffer reusable
    }

    // combine the two half-wave code subsets for the same pixel column
    float oV0 = __shfl_xor(bestV0, 16);
    int   oI0 = __shfl_xor(bestI0, 16);
    if (oV0 > bestV0 || (oV0 == bestV0 && oI0 < bestI0)) { bestV0 = oV0; bestI0 = oI0; }
    float oV1 = __shfl_xor(bestV1, 16);
    int   oI1 = __shfl_xor(bestI1, 16);
    if (oV1 > bestV1 || (oV1 == bestV1 && oI1 < bestI1)) { bestV1 = oV1; bestI1 = oI1; }

    if (lane < 16) {
        int n0 = (blockIdx.x * 8 + wv) * 16 + lane;           // round-0 tile
        int n1 = (blockIdx.x * 8 + 4 + wv) * 16 + lane;       // round-1 tile
        idx_ws[n0]  = bestI0;  idx_out[n0] = bestI0;
        idx_ws[n1]  = bestI1;  idx_out[n1] = bestI1;
    }
}

// ---------------------------------------------------------------------------
// Kernel 3: gather selected codes -> quantized [B,C,H,W] (coalesced along hw),
// accumulate commitment-loss sum of squares.
// ---------------------------------------------------------------------------
__global__ __launch_bounds__(256)
void vq_gather(const float* __restrict__ x, const float* __restrict__ w,
               const int* __restrict__ idx, float* __restrict__ out,
               float* __restrict__ loss_acc) {
    __shared__ float red[256];
    const int n  = blockIdx.x * 256 + threadIdx.x;
    const int b  = n >> 10;
    const int hw = n & 1023;
    const int ci = idx[n];
    const float* wr = w + ((size_t)ci << 8);
    const size_t xb = ((size_t)b << 18) + (size_t)hw;
    float ls = 0.0f;
    for (int c = 0; c < 256; c += 4) {
        float4 w4 = *(const float4*)(wr + c);
        float wa[4] = { w4.x, w4.y, w4.z, w4.w };
        #pragma unroll
        for (int i = 0; i < 4; ++i) {
            size_t off = xb + ((size_t)(c + i) << 10);
            float xv = x[off];
            out[off] = wa[i];
            float d = wa[i] - xv;
            ls += d * d;
        }
    }
    red[threadIdx.x] = ls;
    __syncthreads();
    for (int s = 128; s > 0; s >>= 1) {
        if (threadIdx.x < s) red[threadIdx.x] += red[threadIdx.x + s];
        __syncthreads();
    }
    if (threadIdx.x == 0) atomicAdd(loss_acc, red[0]);
}

// ---------------------------------------------------------------------------
// Kernel 4: finalize scalar loss.
// ---------------------------------------------------------------------------
__global__ void vq_final(const float* __restrict__ loss_acc,
                         const float* __restrict__ commit,
                         float* __restrict__ loss_out) {
    *loss_out = commit[0] * (*loss_acc) / 8388608.0f;  // mean over B*C*H*W
}

extern "C" void kernel_launch(void* const* d_in, const int* in_sizes, int n_in,
                              void* d_out, int out_size, void* d_ws, size_t ws_size,
                              hipStream_t stream) {
    const float* x      = (const float*)d_in[0];   // [32,256,32,32]
    const float* w      = (const float*)d_in[1];   // [2048,256]
    const float* commit = (const float*)d_in[3];   // scalar

    float* out = (float*)d_out;
    float* quant_out = out;                         // 8,388,608 f32
    float* loss_out  = out + 8388608;               // 1 f32
    int*   idx_out   = (int*)(out + 8388609);       // 32,768 i32

    unsigned short* waug = (unsigned short*)d_ws;                    // 1,179,648 B
    int* idx_ws = (int*)((char*)d_ws + (size_t)NE * KAUG * 2);       //   131,072 B
    float* loss_acc = (float*)((char*)d_ws + (size_t)NE * KAUG * 2
                                           + (size_t)NPIX * 4);      //         4 B

    vq_prep  <<<NE,         256, 0, stream>>>(w, waug, loss_acc);
    vq_argmin<<<NPIX / 128, 128, 0, stream>>>(x, waug, idx_ws, idx_out);
    vq_gather<<<NPIX / 256, 256, 0, stream>>>(x, w, idx_ws, quant_out, loss_acc);
    vq_final <<<1,          1,   0, stream>>>(loss_acc, commit, loss_out);
}
